// SVD_Layer_2422361555157
// MI455X (gfx1250) — matrix operand order repaired and checked by reading, not compiled, not run on hardware
//
#include <hip/hip_runtime.h>
#include <hip/hip_bf16.h>

typedef __attribute__((ext_vector_type(16))) __bf16 v16bf;
typedef __attribute__((ext_vector_type(8)))  float  v8f;

#define M_TOK 32768
#define K_DIM 768
#define N_DIM 3072
#define NSV   134
#define IOS   (12*12*NSV)   // 19296

#define BM 128
#define BN 256
#define BK 64
#define KPAD 72             // padded LDS row stride (elements): 144B, 16B-aligned
#define KTILES (K_DIM / BK) // 12

// ---------------- kernel 1: x f32 -> bf16 ----------------
__global__ __launch_bounds__(256)
void cvt_x_kernel(const float* __restrict__ x, __bf16* __restrict__ xb, int n) {
    int idx = blockIdx.x * blockDim.x + threadIdx.x;
    size_t base = (size_t)idx * 8;
    if (base + 8 <= (size_t)n) {
        float4 a = *(const float4*)(x + base);
        float4 c = *(const float4*)(x + base + 4);
        union { __bf16 h[8]; uint4 q; } u;
        u.h[0] = (__bf16)a.x; u.h[1] = (__bf16)a.y;
        u.h[2] = (__bf16)a.z; u.h[3] = (__bf16)a.w;
        u.h[4] = (__bf16)c.x; u.h[5] = (__bf16)c.y;
        u.h[6] = (__bf16)c.z; u.h[7] = (__bf16)c.w;
        *(uint4*)(xb + base) = u.q;
    }
}

// ---------------- kernel 2: T[(i,o),(I,O,S)] = sum_s W1[io,s]*W2[s,(I,O,S)] ----
__global__ __launch_bounds__(256)
void build_T_kernel(const float* __restrict__ W1, const float* __restrict__ W2,
                    float* __restrict__ T) {
    int t  = blockIdx.x * blockDim.x + threadIdx.x;   // 0..19295
    int io = blockIdx.y;                               // 0..127
    if (t >= IOS) return;
    float sum = 0.f;
    const float* w1 = W1 + io * NSV;
    #pragma unroll 2
    for (int s = 0; s < NSV; ++s)
        sum = fmaf(w1[s], W2[(size_t)s * IOS + t], sum);
    T[(size_t)io * IOS + t] = sum;
}

// ---------------- kernel 3: Wt[col][row] = sum_S T[..]*W3[..] (bf16, transposed) ----
__global__ __launch_bounds__(256)
void build_Wt_kernel(const float* __restrict__ T, const float* __restrict__ W3,
                     __bf16* __restrict__ Wt) {
    int col = blockIdx.x * blockDim.x + threadIdx.x;  // 0..3071
    int row = blockIdx.y;                              // 0..767
    if (col >= N_DIM) return;
    int i = row / 96, I = (row / 8) % 12, a = row & 7;
    int o = col / 192, O = (col / 16) % 12, bb = col & 15;
    const float* tr = T + (size_t)(i * 16 + o) * IOS + (size_t)(I * 12 + O) * NSV;
    float sum = 0.f;
    #pragma unroll 2
    for (int S = 0; S < NSV; ++S)
        sum = fmaf(tr[S], W3[(S * 8 + a) * 16 + bb], sum);
    Wt[(size_t)col * K_DIM + row] = (__bf16)sum;
}

// ---------------- kernel 4: out = xb @ Wt^T + b via v_wmma_f32_16x16x32_bf16 ----
// 256 threads = 8 waves in a 2(M) x 4(N) grid; wave tile 64x64 = 4x4 WMMA tiles.
// Double-buffered LDS filled with global_load_async_to_lds_b128 (ASYNCcnt path).
__global__ __launch_bounds__(256)
void gemm_kernel(const __bf16* __restrict__ A,   // M x K (row-major bf16)
                 const __bf16* __restrict__ Bt,  // N x K (row-major bf16 = W^T)
                 const float* __restrict__ bias,
                 float* __restrict__ out) {
    __shared__ __align__(16) __bf16 sA[2][BM * KPAD];   // 2 * 18,432 B
    __shared__ __align__(16) __bf16 sB[2][BN * KPAD];   // 2 * 36,864 B

    const int tid   = threadIdx.x;
    const int lane  = tid & 31;
    const int wave  = tid >> 5;
    const int waveM = wave & 1;    // 2 waves along M (64 rows each)
    const int waveN = wave >> 1;   // 4 waves along N (64 cols each)
    const int l16     = lane & 15;
    const int halfsel = lane >> 4;

    const int blockN = blockIdx.x * BN;
    const int blockM = blockIdx.y * BM;

    v8f acc[4][4];
    #pragma unroll
    for (int mt = 0; mt < 4; ++mt)
        #pragma unroll
        for (int nt = 0; nt < 4; ++nt)
            acc[mt][nt] = (v8f){0.f,0.f,0.f,0.f,0.f,0.f,0.f,0.f};

    // Issue one tile's async copies: A = 128 rows x 64, B = 256 rows x 64,
    // 16B (8-element) segments -> 4 + 8 = 12 async loads per wave per tile.
    auto issueTile = [&](int kt, int buf) {
        #pragma unroll
        for (int p = 0; p < 4; ++p) {
            int s = tid + p * 256;
            int row = s >> 3, seg = (s & 7) * 8;
            const __bf16* g = A + (size_t)(blockM + row) * K_DIM + kt + seg;
            unsigned      l = (unsigned)(uintptr_t)&sA[buf][row * KPAD + seg];
            asm volatile("global_load_async_to_lds_b128 %0, %1, off"
                         :: "v"(l), "v"((unsigned long long)(uintptr_t)g)
                         : "memory");
        }
        #pragma unroll
        for (int p = 0; p < 8; ++p) {
            int s = tid + p * 256;
            int row = s >> 3, seg = (s & 7) * 8;
            const __bf16* g = Bt + (size_t)(blockN + row) * K_DIM + kt + seg;
            unsigned      l = (unsigned)(uintptr_t)&sB[buf][row * KPAD + seg];
            asm volatile("global_load_async_to_lds_b128 %0, %1, off"
                         :: "v"(l), "v"((unsigned long long)(uintptr_t)g)
                         : "memory");
        }
    };

    issueTile(0, 0);

    for (int kt = 0; kt < KTILES; ++kt) {
        const int cur = kt & 1;
        if (kt + 1 < KTILES) {
            issueTile((kt + 1) * BK, cur ^ 1);   // prefetch next tile (DMA)
            // async loads complete in order: <=12 outstanding means tile kt done
            asm volatile("s_wait_asynccnt 12" ::: "memory");
        } else {
            asm volatile("s_wait_asynccnt 0" ::: "memory");
        }
        __syncthreads();

        #pragma unroll
        for (int kk = 0; kk < 2; ++kk) {          // two K=32 steps per tile
            const int k0 = kk * 32;
            // A fragments (16x32 bf16): lanes<16 take K[0..7]+K[16..23],
            // lanes>=16 take K[8..15]+K[24..31]   (ISA 7.12.2 layout)
            union { v16bf v; uint4 q[2]; } af[4];
            #pragma unroll
            for (int mt = 0; mt < 4; ++mt) {
                const __bf16* p0 = &sA[cur][(waveM * 64 + mt * 16 + l16) * KPAD
                                            + k0 + halfsel * 8];
                af[mt].q[0] = *(const uint4*)(p0);
                af[mt].q[1] = *(const uint4*)(p0 + 16);
            }
            // B fragments (32x16 bf16): lanes<16 take K[0..7]+K[16..23], lanes>=16 take K[8..15]+K[24..31]
            union { v16bf v; uint4 q[2]; } bfr[4];
            #pragma unroll
            for (int nt = 0; nt < 4; ++nt) {
                const __bf16* p0 = &sB[cur][(waveN * 64 + nt * 16 + l16) * KPAD
                                            + k0 + halfsel * 8];
                bfr[nt].q[0] = *(const uint4*)(p0);
                bfr[nt].q[1] = *(const uint4*)(p0 + 16);
            }
            #pragma unroll
            for (int mt = 0; mt < 4; ++mt)
                #pragma unroll
                for (int nt = 0; nt < 4; ++nt)
                    acc[mt][nt] = __builtin_amdgcn_wmma_f32_16x16x32_bf16(
                        false, af[mt].v, false, bfr[nt].v,
                        (short)0, acc[mt][nt], false, false);
        }
        __syncthreads();   // protect ping-pong buffer before next DMA issue
    }

    // epilogue: C/D layout: vgpr r -> M = mt*16 + halfsel*8 + r, N = l16
    #pragma unroll
    for (int nt = 0; nt < 4; ++nt) {
        int col = blockN + waveN * 64 + nt * 16 + l16;
        float bv = bias[col];
        #pragma unroll
        for (int mt = 0; mt < 4; ++mt) {
            int row0 = blockM + waveM * 64 + mt * 16 + halfsel * 8;
            #pragma unroll
            for (int r = 0; r < 8; ++r)
                out[(size_t)(row0 + r) * N_DIM + col] = acc[mt][nt][r] + bv;
        }
    }
}

extern "C" void kernel_launch(void* const* d_in, const int* in_sizes, int n_in,
                              void* d_out, int out_size, void* d_ws, size_t ws_size,
                              hipStream_t stream) {
    const float* x  = (const float*)d_in[0];
    const float* W1 = (const float*)d_in[1];
    const float* W2 = (const float*)d_in[2];
    const float* W3 = (const float*)d_in[3];
    const float* b  = (const float*)d_in[4];
    float* out = (float*)d_out;

    // workspace layout (16B aligned)
    char* ws = (char*)d_ws;
    __bf16* xb = (__bf16*)ws;                                  // 32768*768*2  = 50331648 B
    __bf16* Wt = (__bf16*)(ws + (size_t)50331648);             // 3072*768*2   =  4718592 B
    float*  T  = (float*) (ws + (size_t)50331648 + 4718592);   // 128*19296*4  =  9879552 B

    // 1) x -> bf16
    {
        int n = M_TOK * K_DIM;                // 25,165,824 (divisible by 8*256)
        int blocks = n / (8 * 256);
        cvt_x_kernel<<<blocks, 256, 0, stream>>>(x, xb, n);
    }
    // 2) T = W1 (128x134) x W2 (134x19296)
    {
        dim3 grid((IOS + 255) / 256, 128);
        build_T_kernel<<<grid, 256, 0, stream>>>(W1, W2, T);
    }
    // 3) Wt[col][row] = sum_S T * W3  (bf16, stored K-major for the GEMM)
    {
        dim3 grid(N_DIM / 256, K_DIM);
        build_Wt_kernel<<<grid, 256, 0, stream>>>(T, W3, Wt);
    }
    // 4) out = xb @ W + b
    {
        dim3 grid(N_DIM / BN, M_TOK / BM);    // (12, 256)
        gemm_kernel<<<grid, 256, 0, stream>>>(xb, Wt, b, out);
    }
}
